// CausalSelfAttention_635655159877
// MI455X (gfx1250) — matrix operand order repaired and checked by reading, not compiled, not run on hardware
//
#include <hip/hip_runtime.h>

// ---------------------------------------------------------------------------
// CDNA5 (gfx1250) fused causal self-attention:
//   1) convert fp32 -> bf16 (x, w_qkv^T, w_proj^T)
//   2) QKV GEMM  (v_wmma_f32_16x16x32_bf16), tiles staged via TDM
//      (tensor_load_to_lds + s_wait_tensorcnt), scatter to [B,H,T,D] Q/K/V
//   3) flash-style causal attention (QK^T + online softmax + PV, all WMMA;
//      K tile staged via TDM, V tile transposed manually)
//   4) output projection GEMM + bias -> fp32 d_out
// B=2, T=2048, C=1024, H=16, D=64
// ---------------------------------------------------------------------------

typedef unsigned short u16_t;
typedef unsigned int   u32_t;

typedef __attribute__((ext_vector_type(16))) __bf16 v16bf;
typedef __attribute__((ext_vector_type(8)))  float  v8f;
typedef __attribute__((ext_vector_type(4)))  unsigned int u32x4;
typedef __attribute__((ext_vector_type(8)))  unsigned int u32x8;

union Frag { v16bf bf; u32_t u[8]; };

#define NEG_INF   (-__builtin_inff())
#define NEG_BIG   (-1.0e37f)

static __device__ __forceinline__ u16_t f2bf(float x) {
  u32_t u = __float_as_uint(x);
  u32_t r = (u + 0x7FFFu + ((u >> 16) & 1u)) >> 16;   // round-to-nearest-even
  return (u16_t)r;
}

static __device__ __forceinline__ v8f wmma_bf16(const Frag& a, const Frag& b, v8f c) {
  // D = A(16x32 bf16) * B(32x16 bf16) + C(16x16 f32)
  return __builtin_amdgcn_wmma_f32_16x16x32_bf16(
      /*neg_a=*/false, a.bf, /*neg_b=*/false, b.bf,
      /*c_mod=*/(short)0, c, /*reuse_a=*/false, /*reuse_b=*/false);
}

static __device__ __forceinline__ float rmax16(float v) {
  v = fmaxf(v, __shfl_xor(v, 1, 32));
  v = fmaxf(v, __shfl_xor(v, 2, 32));
  v = fmaxf(v, __shfl_xor(v, 4, 32));
  v = fmaxf(v, __shfl_xor(v, 8, 32));   // masks <=8 stay inside each 16-lane half
  return v;
}
static __device__ __forceinline__ float rsum16(float v) {
  v += __shfl_xor(v, 1, 32);
  v += __shfl_xor(v, 2, 32);
  v += __shfl_xor(v, 4, 32);
  v += __shfl_xor(v, 8, 32);
  return v;
}

// ---------------------------------------------------------------------------
// TDM: DMA a 2-D bf16 tile (tile1 rows of tile0 contiguous elements, row
// stride stride0 elements) from global memory into LDS at lds_off bytes.
// D# per CDNA5 ISA; tensor dims set huge so the
// OOB clamp never engages (our tiles always divide the tensors evenly).
// Tracked by TENSORcnt; caller does s_wait_tensorcnt + workgroup barrier.
// ---------------------------------------------------------------------------
static __device__ __forceinline__ void tdm_load_2d(unsigned lds_off,
                                                   const void* gptr,
                                                   unsigned tile0, unsigned tile1,
                                                   unsigned long long stride0) {
  unsigned long long ga = (unsigned long long)(uintptr_t)gptr;
  u32x4 g0;
  g0[0] = 1u;                                   // count=1 valid descriptor
  g0[1] = lds_off;                              // LDS byte address
  g0[2] = (unsigned)ga;                         // global_addr[31:0]
  g0[3] = (unsigned)((ga >> 32) & 0x01FFFFFFull) | (2u << 30); // [56:32]|type=2
  u32x8 g1;
  const unsigned td = 1u << 30;                 // huge tensor dims (no OOB)
  g1[0] = (1u << 16);                           // data_size=1 -> 2 bytes
  g1[1] = (td & 0xFFFFu) << 16;                 // tensor_dim0[15:0]
  g1[2] = ((td >> 16) & 0xFFFFu) | ((td & 0xFFFFu) << 16);   // tdim0 hi | tdim1 lo
  g1[3] = ((td >> 16) & 0xFFFFu) | ((tile0 & 0xFFFFu) << 16);// tdim1 hi | tile_dim0
  g1[4] = tile1 & 0xFFFFu;                      // tile_dim1 (tile_dim2 = 0)
  g1[5] = (unsigned)(stride0 & 0xFFFFFFFFull);  // tensor_dim0_stride[31:0]
  g1[6] = (unsigned)((stride0 >> 32) & 0xFFFFull); // stride[47:32] | tdim1_stride lo (0)
  g1[7] = 0u;
  asm volatile("tensor_load_to_lds %0, %1" :: "s"(g0), "s"(g1) : "memory");
}

// ------------------------------- converters --------------------------------

__global__ void cvt_bf16_kernel(const float* __restrict__ src,
                                u16_t* __restrict__ dst, int n) {
  int i = blockIdx.x * blockDim.x + threadIdx.x;
  if (i < n) dst[i] = f2bf(src[i]);
}

// src is [K][N] fp32 row-major; dst is [N][K] bf16 row-major (transposed)
__global__ void cvt_transpose_kernel(const float* __restrict__ src,
                                     u16_t* __restrict__ dst, int K, int N) {
  int i = blockIdx.x * blockDim.x + threadIdx.x;
  if (i < K * N) {
    int k = i / N, n = i - k * N;
    dst[(size_t)n * K + k] = f2bf(src[i]);
  }
}

// --------------------------------- GEMM ------------------------------------
// C[M,N] = A[M,Kd] * Bt[N,Kd]^T + bias[N]
// Block: 256 threads = 8 waves as 2(M) x 4(N); block tile 64x128;
// wave tile 32x32 = 2x2 WMMA 16x16 tiles; K stepped by 32.
// Tiles staged global->LDS by the Tensor Data Mover (wave 0 issues, all wait).
// MODE 0: scatter bf16 into Q/K/V [B,H,T,D].   MODE 1: fp32 store to Out.

template <int MODE>
__global__ __launch_bounds__(256)
void gemm_bf16_kernel(const u16_t* __restrict__ A,
                      const u16_t* __restrict__ Bt,
                      const float* __restrict__ bias,
                      u16_t* __restrict__ Qb, u16_t* __restrict__ Kb,
                      u16_t* __restrict__ Vb,
                      float* __restrict__ Out, int Kd) {
  __shared__ __align__(16) u16_t As[64][32];
  __shared__ __align__(16) u16_t Bs[128][32];

  const int tid  = threadIdx.x;
  const int wave = tid >> 5;
  const int lane = tid & 31;
  const int half = lane >> 4;
  const int lo   = lane & 15;
  const int wm   = wave >> 2;       // 0..1
  const int wn   = wave & 3;        // 0..3
  const int m0   = blockIdx.y * 64;
  const int n0   = blockIdx.x * 128;

  const unsigned as_off = (unsigned)(uintptr_t)&As[0][0];  // LDS aperture: low
  const unsigned bs_off = (unsigned)(uintptr_t)&Bs[0][0];  // 32 bits = offset

  v8f acc[2][2] = {};

  for (int k0 = 0; k0 < Kd; k0 += 32) {
    __syncthreads();                 // previous iteration's LDS reads done
    if (wave == 0) {
      tdm_load_2d(as_off, &A[(size_t)m0 * Kd + k0], 32, 64, (unsigned long long)Kd);
      tdm_load_2d(bs_off, &Bt[(size_t)n0 * Kd + k0], 32, 128, (unsigned long long)Kd);
    }
    __builtin_amdgcn_s_wait_tensorcnt(0);   // no-op for waves 1..7
    __syncthreads();

    Frag a[2], b[2];
    #pragma unroll
    for (int mi = 0; mi < 2; ++mi) {
      // A frag: lane -> row lo, elem j -> k = half*16 + j
      const u32_t* Ar = (const u32_t*)As[wm * 32 + mi * 16 + lo];
      #pragma unroll
      for (int g = 0; g < 2; ++g)
        #pragma unroll
        for (int p = 0; p < 4; ++p)
          a[mi].u[g * 4 + p] = Ar[half * 8 + g * 4 + p];
    }
    #pragma unroll
    for (int nj = 0; nj < 2; ++nj) {
      // B frag: lane -> col lo, elem j -> k = half*16 + j
      const u32_t* Br = (const u32_t*)Bs[wn * 32 + nj * 16 + lo];
      #pragma unroll
      for (int p = 0; p < 8; ++p)
        b[nj].u[p] = Br[half * 8 + p];
    }
    #pragma unroll
    for (int mi = 0; mi < 2; ++mi)
      #pragma unroll
      for (int nj = 0; nj < 2; ++nj)
        acc[mi][nj] = wmma_bf16(a[mi], b[nj], acc[mi][nj]);
  }

  // Epilogue. C/D layout: elem r -> row (r + 8*half), col = lo.
  #pragma unroll
  for (int mi = 0; mi < 2; ++mi)
    #pragma unroll
    for (int nj = 0; nj < 2; ++nj)
      #pragma unroll
      for (int r = 0; r < 8; ++r) {
        int m = m0 + wm * 32 + mi * 16 + r + 8 * half;
        int n = n0 + wn * 32 + nj * 16 + lo;
        float v = acc[mi][nj][r] + bias[n];
        if (MODE == 0) {
          int which = n >> 10;            // 0:Q 1:K 2:V
          int c = n & 1023;
          int h = c >> 6, d = c & 63;
          int bb = m >> 11, t = m & 2047; // T = 2048
          size_t off = ((size_t)(bb * 16 + h) * 2048 + t) * 64 + d;
          u16_t bv = f2bf(v);
          if (which == 0)      Qb[off] = bv;
          else if (which == 1) Kb[off] = bv;
          else                 Vb[off] = bv;
        } else {
          Out[(size_t)m * 1024 + n] = v;
        }
      }
}

// ------------------------------ attention ----------------------------------
// One block = 4 waves = 64 queries of one (b,h); each wave owns a 16-query
// tile. Stream 32-key blocks: K staged [key][feat] by TDM, V transposed to
// [feat][key] manually. QK^T: 2 WMMAs per 16-key score tile; PV: 4 WMMAs.

__global__ __launch_bounds__(128)
void attn_kernel(const u16_t* __restrict__ Qg, const u16_t* __restrict__ Kg,
                 const u16_t* __restrict__ Vg, u16_t* __restrict__ Yg) {
  __shared__ __align__(16) u16_t Kt[32][64];      // [key][feat]
  __shared__ __align__(16) u16_t Vt[64][32];      // [feat][key]
  __shared__ __align__(16) u16_t Pls[4][16][32];  // per-wave P bounce

  const int tid  = threadIdx.x;
  const int w    = tid >> 5;
  const int lane = tid & 31;
  const int half = lane >> 4;
  const int lo   = lane & 15;
  const int qb   = blockIdx.x & 31;     // T/64 = 32 query blocks
  const int bh   = blockIdx.x >> 5;     // 0..31 (B*H)
  const int q0w  = qb * 64;
  const int q0   = q0w + w * 16;

  const unsigned kt_off = (unsigned)(uintptr_t)&Kt[0][0];

  // Q fragments (D=64 -> two k=32 A-frags), loaded once per wave.
  Frag aq[2];
  {
    const u32_t* Qr = (const u32_t*)(Qg + ((size_t)bh * 2048 + q0 + lo) * 64);
    #pragma unroll
    for (int f = 0; f < 2; ++f)
      #pragma unroll
      for (int g = 0; g < 2; ++g)
        #pragma unroll
        for (int p = 0; p < 4; ++p)
          aq[f].u[g * 4 + p] = Qr[f * 16 + half * 8 + g * 4 + p];
  }

  float mrun[8], lsum[8];
  #pragma unroll
  for (int r = 0; r < 8; ++r) { mrun[r] = NEG_INF; lsum[r] = 0.0f; }
  v8f Oacc[4] = {};

  const int nkb = qb * 2 + 2;          // causal: keys up to q0w+63
  for (int kb = 0; kb < nkb; ++kb) {
    const int kbase = kb * 32;
    __syncthreads();
    // K tile 32x64 via TDM (rows contiguous, row stride D=64)
    if (w == 0)
      tdm_load_2d(kt_off, Kg + ((size_t)bh * 2048 + kbase) * 64, 64, 32, 64ull);
    {
      // V tile transposed manually (TDM cannot transpose)
      int i0 = tid * 2;
      #pragma unroll
      for (int i = i0; i < i0 + 2; ++i) {
        int r = i >> 3, ch = i & 7;
        uint4 vv = *(const uint4*)&Vg[((size_t)bh * 2048 + kbase + r) * 64 + ch * 8];
        const u16_t* pv = (const u16_t*)&vv;
        #pragma unroll
        for (int e = 0; e < 8; ++e) Vt[ch * 8 + e][r] = pv[e];
      }
    }
    __builtin_amdgcn_s_wait_tensorcnt(0);
    __syncthreads();

    // scores: s[kt] = Q(16x64) * K^T(64x16) for key tiles kt*16
    v8f s[2] = {};
    #pragma unroll
    for (int kt = 0; kt < 2; ++kt)
      #pragma unroll
      for (int f = 0; f < 2; ++f) {
        Frag bk;  // B frag: lane -> key lo, elem j -> feat f*32 + half*16 + j
        const u32_t* Kr = (const u32_t*)Kt[kt * 16 + lo];
        #pragma unroll
        for (int p = 0; p < 8; ++p) bk.u[p] = Kr[f * 16 + half * 8 + p];
        s[kt] = wmma_bf16(aq[f], bk, s[kt]);
      }

    // online softmax (row = r + 8*half, col = lane lo within key tile)
    #pragma unroll
    for (int r = 0; r < 8; ++r) {
      int q = q0 + r + 8 * half;
      float v0 = s[0][r] * 0.125f;               // 1/sqrt(64)
      float v1 = s[1][r] * 0.125f;
      if (kbase + lo > q)      v0 = NEG_INF;
      if (kbase + 16 + lo > q) v1 = NEG_INF;
      float mnew = fmaxf(mrun[r], rmax16(fmaxf(v0, v1)));
      float fct = (mrun[r] <= NEG_BIG) ? 0.0f : __expf(mrun[r] - mnew);
      float p0  = (v0 <= NEG_BIG) ? 0.0f : __expf(v0 - mnew);
      float p1  = (v1 <= NEG_BIG) ? 0.0f : __expf(v1 - mnew);
      lsum[r] = lsum[r] * fct + rsum16(p0 + p1);
      mrun[r] = mnew;
      Pls[w][r + 8 * half][lo]      = f2bf(p0);
      Pls[w][r + 8 * half][16 + lo] = f2bf(p1);
      #pragma unroll
      for (int c = 0; c < 4; ++c) Oacc[c][r] *= fct;
    }

    // O += P(16x32) * V(32x64); P re-read from LDS in A-frag layout
    Frag ap;
    {
      const u32_t* Pr = (const u32_t*)Pls[w][lo];
      #pragma unroll
      for (int g = 0; g < 2; ++g)
        #pragma unroll
        for (int p = 0; p < 4; ++p)
          ap.u[g * 4 + p] = Pr[half * 8 + g * 4 + p];
    }
    #pragma unroll
    for (int c = 0; c < 4; ++c) {
      Frag bv;  // lane -> d-col c*16+lo, elem j -> key half*16 + j
      const u32_t* Vr = (const u32_t*)Vt[c * 16 + lo];
      #pragma unroll
      for (int p = 0; p < 8; ++p) bv.u[p] = Vr[half * 8 + p];
      Oacc[c] = wmma_bf16(ap, bv, Oacc[c]);
    }
  }

  // normalize and store y as bf16 [B,T,C] with c = h*64 + d
  const int bb = bh >> 4, h = bh & 15;
  #pragma unroll
  for (int r = 0; r < 8; ++r) {
    float rl = (lsum[r] > 0.0f) ? (1.0f / lsum[r]) : 0.0f;
    int q = q0 + r + 8 * half;
    #pragma unroll
    for (int c = 0; c < 4; ++c) {
      int d = c * 16 + lo;
      Yg[((size_t)bb * 2048 + q) * 1024 + h * 64 + d] = f2bf(Oacc[c][r] * rl);
    }
  }
}

// ------------------------------- launcher ----------------------------------

extern "C" void kernel_launch(void* const* d_in, const int* in_sizes, int n_in,
                              void* d_out, int out_size, void* d_ws, size_t ws_size,
                              hipStream_t stream) {
  (void)in_sizes; (void)n_in; (void)out_size; (void)ws_size;
  const float* x      = (const float*)d_in[0];   // [2,2048,1024]
  const float* w_qkv  = (const float*)d_in[1];   // [1024,3072]
  const float* b_qkv  = (const float*)d_in[2];   // [3072]
  const float* w_proj = (const float*)d_in[3];   // [1024,1024]
  const float* b_proj = (const float*)d_in[4];   // [1024]
  float* out = (float*)d_out;                    // [2,2048,1024] fp32

  u16_t* ws = (u16_t*)d_ws;
  size_t o = 0;
  u16_t* xb     = ws + o; o += (size_t)4096 * 1024;    // x bf16
  u16_t* wqkvT  = ws + o; o += (size_t)3072 * 1024;    // w_qkv^T bf16
  u16_t* wprojT = ws + o; o += (size_t)1024 * 1024;    // w_proj^T bf16
  u16_t* Qb     = ws + o; o += (size_t)32 * 2048 * 64; // [B*H,T,D]
  u16_t* Kb     = ws + o; o += (size_t)32 * 2048 * 64;
  u16_t* Vb     = ws + o; o += (size_t)32 * 2048 * 64;
  u16_t* Yb     = ws + o; o += (size_t)4096 * 1024;    // attn out bf16 [B*T,C]

  cvt_bf16_kernel<<<(4096 * 1024) / 256, 256, 0, stream>>>(x, xb, 4096 * 1024);
  cvt_transpose_kernel<<<(1024 * 3072) / 256, 256, 0, stream>>>(w_qkv, wqkvT, 1024, 3072);
  cvt_transpose_kernel<<<(1024 * 1024) / 256, 256, 0, stream>>>(w_proj, wprojT, 1024, 1024);

  // QKV GEMM: M=4096, N=3072, K=1024
  gemm_bf16_kernel<0><<<dim3(3072 / 128, 4096 / 64), 256, 0, stream>>>(
      xb, wqkvT, b_qkv, Qb, Kb, Vb, nullptr, 1024);

  // attention: grid = B*H * (T/64) = 32 * 32
  attn_kernel<<<1024, 128, 0, stream>>>(Qb, Kb, Vb, Yb);

  // projection GEMM: M=4096, N=1024, K=1024 -> fp32 out (+bias)
  gemm_bf16_kernel<1><<<dim3(1024 / 128, 4096 / 64), 256, 0, stream>>>(
      Yb, wprojT, b_proj, nullptr, nullptr, nullptr, out, 1024);
}
